// WaveNet_76098230551117
// MI455X (gfx1250) — matrix operand order repaired and checked by reading, not compiled, not run on hardware
//
#include <hip/hip_runtime.h>
#include <hip/hip_bf16.h>

typedef __attribute__((ext_vector_type(16))) _Float16 v16h;
typedef __attribute__((ext_vector_type(8)))  _Float16 v8h;
typedef __attribute__((ext_vector_type(8)))  float    v8f;

#define H_CH 128
#define L_FULL 8192
#define NBATCH 4
#define TLEN 6146          // 8192 - 2046
#define RFIELD 2046
#define RSTRIDE 136        // 128 + 8 halves pad (bank-conflict break)
#define ISTRIDE 264        // 256 + 8
#define H1STRIDE 520       // 512 + 8

// ---------------- WMMA helpers ----------------
__device__ __forceinline__ v8f wmma16(v16h a, v16h b, v8f c) {
    // D = A(16x32 f16) * B(32x16 f16) + C(16x16 f32)
    return __builtin_amdgcn_wmma_f32_16x16x32_f16(false, a, false, b, (short)0, c, false, false);
}

// Load a 32x16 f16 B-fragment from a time-major LDS tile buf[t][c] (row stride in halves).
// This code's contiguous k order (matches A repack): lanes 0-15 carry K=0..15 (halves j==K), lanes 16-31 carry K=16..31; N = lane&15.
__device__ __forceinline__ v16h ldsB(const _Float16* buf, int t, int cbase, int stride) {
    const _Float16* p = buf + t * stride + cbase;   // cbase includes +16 for hi half-wave
    v16h r;
    *(v8h*)&r       = *(const v8h*)p;
    *((v8h*)&r + 1) = *(const v8h*)(p + 8);
    return r;
}

// ---------------- weight repack ----------------
// Fragment buffer layout (units of 512 halves per fragment, 32 lanes x 16 halves):
//   [0,64)            : input_w  (8 Mtiles x 8 Ktiles, frag = m*8+k)
//   [64, 64+20*192)   : per layer: F0[32] F1[32] G0[32] G1[32] R[32] S[32], frag-in-mat = m*4+k
//   [3904, 4032)      : fc1 (32 Mtiles x 4 Ktiles)
//   [4032, 4288)      : fc2 (16 Mtiles x 16 Ktiles)
__global__ void wn_prep_weights(const float* win, const float* wf, const float* wg,
                                const float* wr, const float* wsp, const float* wfc1,
                                const float* wfc2, _Float16* out) {
    int f = blockIdx.x;
    int lane = threadIdx.x;
    int lo = lane & 15, hi = lane >> 4;
    _Float16* dst = out + (size_t)f * 512 + lane * 16;
    for (int j = 0; j < 16; ++j) {
        // A-fragment, this code's contiguous k order: K = j + 16*(lane>>4), M = lane&15
        int ck = j + 16 * hi;
        float v;
        if (f < 64) {
            int m = f >> 3, k = f & 7;
            v = win[(m * 16 + lo) * 256 + k * 32 + ck];
        } else if (f < 3904) {
            int g = f - 64;
            int l = g / 192, r = g % 192, mat = r / 32, fi = r % 32;
            int m = fi >> 2, k = fi & 3;
            size_t e = ((size_t)(l * H_CH + m * 16 + lo) * H_CH + k * 32 + ck);
            if      (mat == 0) v = wf[e * 2 + 0];
            else if (mat == 1) v = wf[e * 2 + 1];
            else if (mat == 2) v = wg[e * 2 + 0];
            else if (mat == 3) v = wg[e * 2 + 1];
            else if (mat == 4) v = wr[e];
            else               v = wsp[e];
        } else if (f < 4032) {
            int g = f - 3904; int m = g >> 2, k = g & 3;
            v = wfc1[(m * 16 + lo) * 128 + k * 32 + ck];
        } else {
            int g = f - 4032; int m = g >> 4, k = g & 15;
            v = wfc2[(m * 16 + lo) * 512 + k * 32 + ck];
        }
        dst[j] = (_Float16)v;
    }
}

__global__ void wn_zero(float* p, long n) {
    long i = (long)blockIdx.x * blockDim.x + threadIdx.x;
    if (i < n) p[i] = 0.f;
}

// ---------------- input 1x1 conv: r = input_w @ inputs ----------------
__global__ __launch_bounds__(256)
void wn_input_conv(const float* __restrict__ inp, const _Float16* __restrict__ wfrag,
                   float* __restrict__ rdst) {
    __shared__ _Float16 xb[64 * ISTRIDE];
    int n = blockIdx.y, t0 = blockIdx.x * 64, tid = threadIdx.x;
    const float* in_n = inp + (size_t)n * 256 * L_FULL;
    for (int idx = tid; idx < 64 * 256; idx += 256) {
        int t = idx & 63, c = idx >> 6;
        xb[t * ISTRIDE + c] = (_Float16)in_n[(size_t)c * L_FULL + t0 + t];
    }
    __syncthreads();
    int w = tid >> 5, lane = tid & 31, lo = lane & 15, hi = lane >> 4;
    v8f acc[4] = {};
    for (int ks = 0; ks < 8; ++ks) {
        v16h a = *(const v16h*)(wfrag + ((size_t)(w * 8 + ks) * 512 + lane * 16));
        int cb = ks * 32 + 16 * hi;
        for (int nt = 0; nt < 4; ++nt)
            acc[nt] = wmma16(a, ldsB(xb, nt * 16 + lo, cb, ISTRIDE), acc[nt]);
    }
    float* rn = rdst + (size_t)n * H_CH * L_FULL;
    for (int nt = 0; nt < 4; ++nt) {
        int t = t0 + nt * 16 + lo;
        for (int v = 0; v < 8; ++v)
            rn[(size_t)(w * 16 + v + 8 * hi) * L_FULL + t] = acc[nt][v];
    }
}

// ---------------- fused residual layer ----------------
__global__ __launch_bounds__(256)
void wn_layer(const float* __restrict__ rcur, float* __restrict__ rnext,
              float* __restrict__ finalbuf, const _Float16* __restrict__ wfrag,
              int d, int off, int Lout) {
    __shared__ _Float16 x0[64 * RSTRIDE];
    __shared__ _Float16 x1[64 * RSTRIDE];
    __shared__ _Float16 ob[64 * RSTRIDE];
    int n = blockIdx.y, t0 = blockIdx.x * 64, tid = threadIdx.x;
    const float* rn = rcur + (size_t)n * H_CH * L_FULL;
    for (int idx = tid; idx < 64 * 128; idx += 256) {
        int t = idx & 63, c = idx >> 6;
        int p0 = off + t0 + t, p1 = p0 + d;
        float a = (p0 < L_FULL) ? rn[(size_t)c * L_FULL + p0] : 0.f;
        float b = (p1 < L_FULL) ? rn[(size_t)c * L_FULL + p1] : 0.f;
        x0[t * RSTRIDE + c] = (_Float16)a;
        x1[t * RSTRIDE + c] = (_Float16)b;
    }
    __syncthreads();
    int w = tid >> 5, lane = tid & 31, lo = lane & 15, hi = lane >> 4;
    v8f accF[4] = {}, accG[4] = {};
    for (int ks = 0; ks < 4; ++ks) {
        int cb = ks * 32 + 16 * hi;
        size_t lidx = (size_t)(w * 4 + ks) * 512 + lane * 16;
        v16h aF = *(const v16h*)(wfrag + 0 * 32 * 512 + lidx);   // filter tap0
        v16h aG = *(const v16h*)(wfrag + 2 * 32 * 512 + lidx);   // gate tap0
        for (int nt = 0; nt < 4; ++nt) {
            v16h b = ldsB(x0, nt * 16 + lo, cb, RSTRIDE);
            accF[nt] = wmma16(aF, b, accF[nt]);
            accG[nt] = wmma16(aG, b, accG[nt]);
        }
        aF = *(const v16h*)(wfrag + 1 * 32 * 512 + lidx);        // filter tap1
        aG = *(const v16h*)(wfrag + 3 * 32 * 512 + lidx);        // gate tap1
        for (int nt = 0; nt < 4; ++nt) {
            v16h b = ldsB(x1, nt * 16 + lo, cb, RSTRIDE);
            accF[nt] = wmma16(aF, b, accF[nt]);
            accG[nt] = wmma16(aG, b, accG[nt]);
        }
    }
    // out = tanh(f) * sigmoid(g) -> LDS
    for (int nt = 0; nt < 4; ++nt) {
        int t = nt * 16 + lo;
        for (int v = 0; v < 8; ++v) {
            int c = w * 16 + v + 8 * hi;
            float o = tanhf(accF[nt][v]) * (1.f / (1.f + __expf(-accG[nt][v])));
            ob[t * RSTRIDE + c] = (_Float16)o;
        }
    }
    __syncthreads();
    v8f accR[4] = {}, accS[4] = {};
    for (int ks = 0; ks < 4; ++ks) {
        int cb = ks * 32 + 16 * hi;
        size_t lidx = (size_t)(w * 4 + ks) * 512 + lane * 16;
        v16h aR = *(const v16h*)(wfrag + 4 * 32 * 512 + lidx);   // res
        v16h aS = *(const v16h*)(wfrag + 5 * 32 * 512 + lidx);   // split/skip
        for (int nt = 0; nt < 4; ++nt) {
            v16h b = ldsB(ob, nt * 16 + lo, cb, RSTRIDE);
            accR[nt] = wmma16(aR, b, accR[nt]);
            accS[nt] = wmma16(aS, b, accS[nt]);
        }
    }
    float* rdn = rnext + (size_t)n * H_CH * L_FULL;
    float* fin = finalbuf + (size_t)n * H_CH * TLEN;
    for (int nt = 0; nt < 4; ++nt) {
        int t = nt * 16 + lo;
        if (t0 + t < Lout) {
            int g = off + d + t0 + t;       // global time position of output t
            int pf = g - RFIELD;
            for (int v = 0; v < 8; ++v) {
                int c = w * 16 + v + 8 * hi;
                rdn[(size_t)c * L_FULL + g] = accR[nt][v] + rn[(size_t)c * L_FULL + g];
                if (pf >= 0) fin[(size_t)c * TLEN + pf] += accS[nt][v];
            }
        }
    }
}

// ---------------- head: relu -> fc1 -> relu -> fc2 -> transpose ----------------
__global__ __launch_bounds__(256)
void wn_head(const float* __restrict__ finalbuf, const _Float16* __restrict__ w1f,
             const _Float16* __restrict__ w2f, const float* __restrict__ b1,
             const float* __restrict__ b2, float* __restrict__ out) {
    __shared__ _Float16 xb[32 * RSTRIDE];
    __shared__ _Float16 h1[32 * H1STRIDE];
    int n = blockIdx.y, t0 = blockIdx.x * 32, tid = threadIdx.x;
    const float* fn = finalbuf + (size_t)n * H_CH * TLEN;
    for (int idx = tid; idx < 32 * 128; idx += 256) {
        int t = idx & 31, c = idx >> 5;
        int tf = t0 + t;
        float v = (tf < TLEN) ? fn[(size_t)c * TLEN + tf] : 0.f;
        xb[t * RSTRIDE + c] = (_Float16)(v > 0.f ? v : 0.f);
    }
    __syncthreads();
    int w = tid >> 5, lane = tid & 31, lo = lane & 15, hi = lane >> 4;
    v8f acc1[4][2] = {};
    for (int ks = 0; ks < 4; ++ks) {
        int cb = ks * 32 + 16 * hi;
        v16h bf0 = ldsB(xb, lo, cb, RSTRIDE);
        v16h bf1 = ldsB(xb, 16 + lo, cb, RSTRIDE);
        for (int q = 0; q < 4; ++q) {
            v16h a = *(const v16h*)(w1f + ((size_t)((w * 4 + q) * 4 + ks) * 512 + lane * 16));
            acc1[q][0] = wmma16(a, bf0, acc1[q][0]);
            acc1[q][1] = wmma16(a, bf1, acc1[q][1]);
        }
    }
    for (int q = 0; q < 4; ++q)
        for (int nt = 0; nt < 2; ++nt) {
            int t = nt * 16 + lo;
            for (int v = 0; v < 8; ++v) {
                int oc = (w * 4 + q) * 16 + v + 8 * hi;
                float x = acc1[q][nt][v] + b1[oc];
                h1[t * H1STRIDE + oc] = (_Float16)(x > 0.f ? x : 0.f);
            }
        }
    __syncthreads();
    v8f acc2[2] = {};
    for (int ks = 0; ks < 16; ++ks) {
        int cb = ks * 32 + 16 * hi;
        v16h a = *(const v16h*)(w2f + ((size_t)(w * 16 + ks) * 512 + lane * 16));
        acc2[0] = wmma16(a, ldsB(h1, lo, cb, H1STRIDE), acc2[0]);
        acc2[1] = wmma16(a, ldsB(h1, 16 + lo, cb, H1STRIDE), acc2[1]);
    }
    for (int nt = 0; nt < 2; ++nt) {
        int tf = t0 + nt * 16 + lo;
        if (tf < TLEN) {
            for (int v = 0; v < 8; ++v) {
                int oc = w * 16 + v + 8 * hi;
                out[((size_t)n * TLEN + tf) * 256 + oc] = acc2[nt][v] + b2[oc];
            }
        }
    }
}

extern "C" void kernel_launch(void* const* d_in, const int* in_sizes, int n_in,
                              void* d_out, int out_size, void* d_ws, size_t ws_size,
                              hipStream_t stream) {
    const float* inputs   = (const float*)d_in[0];
    const float* input_w  = (const float*)d_in[1];
    const float* filter_w = (const float*)d_in[2];
    const float* gate_w   = (const float*)d_in[3];
    const float* res_w    = (const float*)d_in[4];
    const float* split_w  = (const float*)d_in[5];
    const float* fc1_w    = (const float*)d_in[6];
    const float* fc1_b    = (const float*)d_in[7];
    const float* fc2_w    = (const float*)d_in[8];
    const float* fc2_b    = (const float*)d_in[9];
    float* out = (float*)d_out;

    char* ws = (char*)d_ws;
    _Float16* wfrag = (_Float16*)ws;                 // 4288 frags * 1024B = 4,390,912 B
    size_t o = (size_t)4288 * 512 * 2;
    o = (o + 255) & ~(size_t)255;
    float* rA  = (float*)(ws + o); o += (size_t)NBATCH * H_CH * L_FULL * 4;  // 16 MB
    float* rB  = (float*)(ws + o); o += (size_t)NBATCH * H_CH * L_FULL * 4;  // 16 MB
    float* fin = (float*)(ws + o);                                           // 12.6 MB
    (void)ws_size; (void)in_sizes; (void)n_in; (void)out_size;

    // 1) repack weights to f16 WMMA A-fragments
    wn_prep_weights<<<4288, 32, 0, stream>>>(input_w, filter_w, gate_w, res_w,
                                             split_w, fc1_w, fc2_w, wfrag);
    // 2) zero the skip accumulator
    long nf = (long)NBATCH * H_CH * TLEN;
    wn_zero<<<(int)((nf + 255) / 256), 256, 0, stream>>>(fin, nf);
    // 3) input 1x1 conv
    wn_input_conv<<<dim3(L_FULL / 64, NBATCH), 256, 0, stream>>>(inputs, wfrag, rA);
    // 4) 20 fused residual layers (ping-pong r buffers)
    float* rc = rA; float* rx = rB;
    int off = 0;
    for (int l = 0; l < 20; ++l) {
        int d = 1 << (l % 10);
        int Lin = L_FULL - off;
        int Lout = Lin - d;
        int nblk = (Lout + 63) / 64;
        const _Float16* lw = wfrag + ((size_t)64 + (size_t)l * 192) * 512;
        wn_layer<<<dim3(nblk, NBATCH), 256, 0, stream>>>(rc, rx, fin, lw, d, off, Lout);
        off += d;
        float* t = rc; rc = rx; rx = t;
    }
    // 5) head
    const _Float16* w1f = wfrag + (size_t)3904 * 512;
    const _Float16* w2f = wfrag + (size_t)4032 * 512;
    wn_head<<<dim3((TLEN + 31) / 32, NBATCH), 256, 0, stream>>>(fin, w1f, w2f,
                                                               fc1_b, fc2_b, out);
}
